// VGAE_25331717112182
// MI455X (gfx1250) — matrix operand order repaired and checked by reading, not compiled, not run on hardware
//
#include <hip/hip_runtime.h>
#include <hip/hip_bf16.h>

typedef __attribute__((ext_vector_type(16))) _Float16 v16h;
typedef __attribute__((ext_vector_type(8)))  _Float16 v8h;
typedef __attribute__((ext_vector_type(4)))  _Float16 v4h;
typedef __attribute__((ext_vector_type(8)))  float    v8f;
typedef __attribute__((ext_vector_type(4)))  float    v4f;
typedef __attribute__((ext_vector_type(2)))  float    v2f;

// ---------- degree / norm ----------
__global__ void k_fill1(float* deg, int n) {
    int i = blockIdx.x * blockDim.x + threadIdx.x;
    if (i < n) deg[i] = 1.0f;                 // self-loop contributes 1
}

__global__ void k_deg_accum(float* deg, const int* __restrict__ dst, int E) {
    int i = blockIdx.x * blockDim.x + threadIdx.x;
    if (i < E) atomicAdd(&deg[dst[i]], 1.0f);
}

__global__ void k_rsqrt_inplace(float* d, int n) {
    int i = blockIdx.x * blockDim.x + threadIdx.x;
    if (i < n) d[i] = rsqrtf(d[i]);           // deg >= 1 always
}

// ---------- fp32 -> fp16 convert, 4 elements / thread ----------
__global__ void k_cvt_f16x4(const float* __restrict__ src, _Float16* __restrict__ dst, int n4) {
    int i = blockIdx.x * blockDim.x + threadIdx.x;
    if (i >= n4) return;
    v4f s = *(const v4f*)(src + (size_t)i * 4);
    v4h h = { (_Float16)s.x, (_Float16)s.y, (_Float16)s.z, (_Float16)s.w };
    *(v4h*)(dst + (size_t)i * 4) = h;
}

// W is [K, N] row-major fp32; produce Wt [N, K] fp16 (so B-fragment loads are contiguous)
__global__ void k_cvt_transpose_f16(const float* __restrict__ W, _Float16* __restrict__ Wt,
                                    int K, int N) {
    int i = blockIdx.x * blockDim.x + threadIdx.x;
    if (i >= K * N) return;
    int k = i / N, n = i % N;
    Wt[(size_t)n * K + k] = (_Float16)W[i];
}

// ---------- WMMA GEMM: C[M,N] = A[M,128] (f16) @ Bt[N,128]^T (f16), fp32 acc ----------
// One wave per 16x16 output tile, K=128 in four 16x16x32 WMMAs. EXEC all-ones by construction.
__global__ __launch_bounds__(32) void k_wmma_gemm_k128(
    const _Float16* __restrict__ A, const _Float16* __restrict__ Bt,
    float* __restrict__ C, int N) {
    const int K = 128;
    int lane = threadIdx.x & 31;
    int g    = lane >> 4;     // half-wave group
    int mr   = lane & 15;     // row (A) / column (B,D) within tile
    int tm   = blockIdx.x;    // 16-row tile
    int tn   = blockIdx.y;    // 16-col tile

    const _Float16* arow = A  + (size_t)(tm * 16 + mr) * K;
    const _Float16* bcol = Bt + (size_t)(tn * 16 + mr) * K;

    v8f acc = {};
#pragma unroll
    for (int k0 = 0; k0 < K; k0 += 32) {
        // A 16x32 f16 fragment: halfs e -> K = k0 + 16g + e (contiguous)
        v16h a = *(const v16h*)(arow + k0 + 16 * g);
        // B 32x16 f16 fragment: halfs e -> K = k0 + 16g + e (contiguous, W stored transposed)
        v16h b = *(const v16h*)(bcol + k0 + 16 * g);
        acc = __builtin_amdgcn_wmma_f32_16x16x32_f16(false, a, false, b,
                                                     (short)0, acc, false, false);
    }
    // D layout: VGPR r -> row (r + 8g), lane&15 -> column
    float* crow = C + (size_t)(tm * 16 + 8 * g) * N + tn * 16 + mr;
#pragma unroll
    for (int r = 0; r < 8; ++r) crow[(size_t)r * N] = acc[r];
}

// ---------- aggregation ----------
// self-loop term also initializes the accumulator: out[i,:] = dinv[i]^2 * h[i,:]
// 4 floats per thread; F is a multiple of 4 so a thread never crosses a node row.
__global__ void k_selfloop_init4(const float* __restrict__ h, const float* __restrict__ dinv,
                                 float* __restrict__ out, int n4, int F) {
    int i = blockIdx.x * blockDim.x + threadIdx.x;
    if (i >= n4) return;
    int node = (i * 4) / F;
    float w = dinv[node];
    w *= w;
    v4f hv = *(const v4f*)(h + (size_t)i * 4);
    v4f o  = { w * hv.x, w * hv.y, w * hv.z, w * hv.w };
    *(v4f*)(out + (size_t)i * 4) = o;
}

// edge scatter, one wave32 per edge: indices forced to SGPRs, each lane moves a
// contiguous VEC-float chunk (b128/b64 gather) then VEC global_atomic_add_f32.
template <int VEC>
__global__ void k_edge_scatter_wave(const float* __restrict__ h, const float* __restrict__ dinv,
                                    const int* __restrict__ src, const int* __restrict__ dst,
                                    float* __restrict__ out, int E) {
    const int F = 32 * VEC;
    int wid  = (blockIdx.x * blockDim.x + threadIdx.x) >> 5;   // edge id (wave-uniform)
    int lane = threadIdx.x & 31;
    if (wid >= E) return;
    int s = __builtin_amdgcn_readfirstlane(src[wid]);
    int d = __builtin_amdgcn_readfirstlane(dst[wid]);
    float w = dinv[s] * dinv[d];                               // scalar operands
    using vecT = float __attribute__((ext_vector_type(VEC)));
    vecT hv = *(const vecT*)(h + (size_t)s * F + lane * VEC);
    float* od = out + (size_t)d * F + lane * VEC;
#pragma unroll
    for (int i = 0; i < VEC; ++i) atomicAdd(&od[i], w * hv[i]);
}

// layer-1 epilogue: bias + ReLU, emit fp16 for the next WMMA GEMM
__global__ void k_bias_relu_cvt(const float* __restrict__ acc, const float* __restrict__ b,
                                _Float16* __restrict__ out, int n, int F) {
    int i = blockIdx.x * blockDim.x + threadIdx.x;
    if (i >= n * F) return;
    float v = acc[i] + b[i % F];
    out[i] = (_Float16)(v > 0.0f ? v : 0.0f);
}

// layer-2 epilogue: bias only (no ReLU), in place
__global__ void k_bias(float* lat, const float* __restrict__ b, int n, int F) {
    int i = blockIdx.x * blockDim.x + threadIdx.x;
    if (i >= n * F) return;
    lat[i] += b[i % F];
}

// ---------- decoder: one wave32 per edge, 64-dim dot + shuffle reduce ----------
__global__ void k_decode(const float* __restrict__ lat,
                         const int* __restrict__ pos, const int* __restrict__ neg,
                         int Ep, int En, float* __restrict__ out) {
    int w    = (blockIdx.x * blockDim.x + threadIdx.x) >> 5;   // edge id (wave-uniform)
    int lane = threadIdx.x & 31;
    if (w >= Ep + En) return;
    int a, b;
    if (w < Ep) { a = pos[w];      b = pos[Ep + w]; }
    else        { int j = w - Ep;  a = neg[j]; b = neg[En + j]; }
    a = __builtin_amdgcn_readfirstlane(a);
    b = __builtin_amdgcn_readfirstlane(b);
    const float* la = lat + (size_t)a * 64;
    const float* lb = lat + (size_t)b * 64;
    v2f va = *(const v2f*)(la + lane * 2);
    v2f vb = *(const v2f*)(lb + lane * 2);
    float p = va.x * vb.x + va.y * vb.y;
#pragma unroll
    for (int m = 16; m > 0; m >>= 1) p += __shfl_xor(p, m, 32);
    if (lane == 0) out[w] = p;
}

extern "C" void kernel_launch(void* const* d_in, const int* in_sizes, int n_in,
                              void* d_out, int out_size, void* d_ws, size_t ws_size,
                              hipStream_t stream) {
    const float* x  = (const float*)d_in[0];
    const int*   ei = (const int*)d_in[1];   // [2, E] flat: src row then dst row
    const int*   pe = (const int*)d_in[2];
    const int*   ne = (const int*)d_in[3];
    const float* W1 = (const float*)d_in[4]; // [128,128]
    const float* b1 = (const float*)d_in[5];
    const float* W2 = (const float*)d_in[6]; // [128,64]
    const float* b2 = (const float*)d_in[7];

    const int N  = 50000;
    const int K  = 128;
    const int F1 = 128, F2 = 64;
    const int E  = in_sizes[1] / 2;
    const int Ep = in_sizes[2] / 2;
    const int En = in_sizes[3] / 2;

    // ---- workspace carve-up (256B-aligned slabs, ~64.3 MB total) ----
    char* ws = (char*)d_ws;
    size_t off = 0;
    auto take = [&](size_t bytes) -> void* {
        void* p = ws + off;
        off = (off + bytes + 255) & ~(size_t)255;
        return p;
    };
    float*    dinv = (float*)   take((size_t)N * 4);
    _Float16* xh   = (_Float16*)take((size_t)N * K * 2);   // reused as h1 fp16 later
    _Float16* w1t  = (_Float16*)take((size_t)F1 * K * 2);
    _Float16* w2t  = (_Float16*)take((size_t)F2 * K * 2);
    float*    h1   = (float*)   take((size_t)N * F1 * 4);  // reused for h2 + latent later
    float*    out1 = (float*)   take((size_t)N * F1 * 4);
    _Float16* h1h  = xh;                    // x_f16 dead after GEMM1
    float*    h2   = h1;                    // h1 f32 dead after scatter1 epilogue
    float*    lat  = h1 + (size_t)N * F2;   // second half of the h1 slab

    const int T = 256;
    auto B = [](long long n, int t) { return (unsigned)((n + t - 1) / t); };

    // degrees (shared by both layers)
    k_fill1        <<<B(N, T), T, 0, stream>>>(dinv, N);
    k_deg_accum    <<<B(E, T), T, 0, stream>>>(dinv, ei + E, E);
    k_rsqrt_inplace<<<B(N, T), T, 0, stream>>>(dinv, N);

    // fp16 operands
    k_cvt_f16x4        <<<B((long long)N * K / 4, T), T, 0, stream>>>(x, xh, N * K / 4);
    k_cvt_transpose_f16<<<B(K * F1, T), T, 0, stream>>>(W1, w1t, K, F1);
    k_cvt_transpose_f16<<<B(K * F2, T), T, 0, stream>>>(W2, w2t, K, F2);

    // layer 1: GEMM -> self-loop init -> edge scatter (wave/edge) -> bias+ReLU (emit fp16)
    dim3 g1(N / 16, F1 / 16);
    k_wmma_gemm_k128     <<<g1, 32, 0, stream>>>(xh, w1t, h1, F1);
    k_selfloop_init4     <<<B((long long)N * F1 / 4, T), T, 0, stream>>>(h1, dinv, out1, N * F1 / 4, F1);
    k_edge_scatter_wave<4><<<B((long long)E * 32, T), T, 0, stream>>>(h1, dinv, ei, ei + E, out1, E);
    k_bias_relu_cvt      <<<B((long long)N * F1, T), T, 0, stream>>>(out1, b1, h1h, N, F1);

    // layer 2: GEMM -> self-loop init -> edge scatter -> bias
    dim3 g2(N / 16, F2 / 16);
    k_wmma_gemm_k128     <<<g2, 32, 0, stream>>>(h1h, w2t, h2, F2);
    k_selfloop_init4     <<<B((long long)N * F2 / 4, T), T, 0, stream>>>(h2, dinv, lat, N * F2 / 4, F2);
    k_edge_scatter_wave<2><<<B((long long)E * 32, T), T, 0, stream>>>(h2, dinv, ei, ei + E, lat, E);
    k_bias               <<<B((long long)N * F2, T), T, 0, stream>>>(lat, b2, N, F2);

    // decoder
    k_decode<<<B((long long)(Ep + En) * 32, T), T, 0, stream>>>(lat, pe, ne, Ep, En, (float*)d_out);
}
